// GATConv_35296041238791
// MI455X (gfx1250) — matrix operand order repaired and checked by reading, not compiled, not run on hardware
//
#include <hip/hip_runtime.h>
#include <math.h>

// ---- problem constants (from reference) ----
#define IN_FEATS 128
#define HEADS 4
#define DHEAD 16
#define HD 64            // HEADS*DHEAD
#define NEG_SLOPE 0.2f
#define MT 8             // M tiles (of 16 rows) per block

typedef _Float16 v8h  __attribute__((ext_vector_type(8)));
typedef _Float16 v16h __attribute__((ext_vector_type(16)));
typedef float    v8f  __attribute__((ext_vector_type(8)));

union uv16 { v16h v; v8h h[2]; };

// order-preserving float->int key so we can use native signed int atomicMax
__device__ __forceinline__ int orderedKey(float f) {
    int i = __float_as_int(f);
    return (i >= 0) ? i : (i ^ 0x7FFFFFFF);
}
__device__ __forceinline__ float keyToFloat(int k) {
    return __int_as_float((k >= 0) ? k : (k ^ 0x7FFFFFFF));
}

// ---------------- init: zero output, -inf max keys, zero denoms ----------------
__global__ void gat_init(float* __restrict__ out, int outN,
                         int* __restrict__ emax, float* __restrict__ denom, int nh) {
    int i = blockIdx.x * blockDim.x + threadIdx.x;
    if (i < outN) out[i] = 0.0f;
    if (i < nh) { emax[i] = (int)0x80000000u; denom[i] = 0.0f; }
}

// ---- ft = feat @ W via v_wmma_f32_16x16x32_f16, fused el/er epilogue ----
// block = 128 threads = 4 waves. Each wave owns one 16-col tile == one head.
// Block processes MT consecutive 16-row tiles; W fragments staged once and
// kept in registers; A tile double-buffered in LDS (one barrier per tile).
// All 4 A fragments are pulled into registers before the WMMA burst so the
// matrix ops retire against partial dscnt waits instead of serializing.
#define LDS_STRIDE 136   // 128 + 8 halves pad to break bank conflicts
__global__ __launch_bounds__(128) void gat_gemm_wmma(
        const float* __restrict__ feat, const float* __restrict__ W,
        const float* __restrict__ attn_l, const float* __restrict__ attn_r,
        float* __restrict__ ft, float* __restrict__ el, float* __restrict__ er,
        int nnodes) {
    __shared__ _Float16 Bs[64 * LDS_STRIDE];       // W transposed: [n][k]
    __shared__ _Float16 As[2][16 * LDS_STRIDE];    // double-buffered feat tile

    const int tid     = threadIdx.x;
    const int blockM0 = blockIdx.x * (MT * 16);

    // stage W (transposed) into LDS as f16, coalesced global reads
    for (int e = tid; e < IN_FEATS * HD; e += 128) {
        int k = e >> 6, n = e & 63;                // W is [128][64] row-major
        Bs[n * LDS_STRIDE + k] = (_Float16)W[e];
    }

    const int wave = tid >> 5;
    const int lane = tid & 31;
    const int half = lane >> 4;      // 0: lanes 0-15, 1: lanes 16-31
    const int l16  = lane & 15;
    const int n0   = wave * 16;      // wave w covers columns of head w
    const int n    = n0 + l16;

    const float al = attn_l[n];      // attn_l[h=wave][d=l16]
    const float ar = attn_r[n];

    __syncthreads();                 // Bs ready

    // B 32x16 f16 fragments: col N=n; half 0 -> K {kb..+7, kb+16..+23}, half 1 -> +8.
    // Invariant across M tiles -> keep in registers (4 x 8 VGPRs).
    uv16 b[4];
    #pragma unroll
    for (int ks = 0; ks < 4; ++ks) {
        const _Float16* bp = &Bs[n * LDS_STRIDE + ks * 32 + half * 8];
        b[ks].h[0] = *(const v8h*)(bp);        // aligned ds_load_b128
        b[ks].h[1] = *(const v8h*)(bp + 16);
    }

    for (int t = 0; t < MT; ++t) {
        const int m0 = blockM0 + t * 16;
        if (m0 >= nnodes) break;               // block-uniform

        _Float16* A = As[t & 1];
        for (int e = tid; e < 16 * IN_FEATS; e += 128) {
            int r = e >> 7, k = e & 127;
            int row = m0 + r;
            float v = (row < nnodes) ? feat[(size_t)row * IN_FEATS + k] : 0.0f;
            A[r * LDS_STRIDE + k] = (_Float16)v;
        }
        __syncthreads();

        // pull all 4 A fragments (8x ds_load_b128) before the WMMA burst
        uv16 a[4];
        #pragma unroll
        for (int ks = 0; ks < 4; ++ks) {
            // A 16x32 f16: row M=l16; half 0 -> K {kb..+7, kb+16..+23}, half 1 -> +8
            const _Float16* ap = &A[l16 * LDS_STRIDE + ks * 32 + half * 8];
            a[ks].h[0] = *(const v8h*)(ap);
            a[ks].h[1] = *(const v8h*)(ap + 16);
        }

        v8f acc = {};
        #pragma unroll
        for (int ks = 0; ks < 4; ++ks)
            acc = __builtin_amdgcn_wmma_f32_16x16x32_f16(
                      false, a[ks].v, false, b[ks].v, (short)0, acc, false, false);

        // C/D layout: VGPR r -> M = m0 + r + half*8, N = n. Coalesced across lanes.
        if (m0 + 16 <= nnodes) {               // block-uniform fast path
            #pragma unroll
            for (int r = 0; r < 8; ++r)
                ft[(size_t)(m0 + r + half * 8) * HD + n] = acc[r];
        } else {
            #pragma unroll
            for (int r = 0; r < 8; ++r) {
                int m = m0 + r + half * 8;
                if (m < nnodes) ft[(size_t)m * HD + n] = acc[r];
            }
        }

        // fused el/er: 16-lane butterfly reduction over d within each half-wave
        #pragma unroll
        for (int r = 0; r < 8; ++r) {
            float sl = acc[r] * al;
            float sr = acc[r] * ar;
            #pragma unroll
            for (int m = 8; m >= 1; m >>= 1) {
                sl += __shfl_xor(sl, m, 32);
                sr += __shfl_xor(sr, m, 32);
            }
            int row = m0 + r + half * 8;
            if (l16 == 0 && row < nnodes) {
                el[row * HEADS + wave] = sl;
                er[row * HEADS + wave] = sr;
            }
        }
        // double buffer: next iteration stages As[(t+1)&1]; As[t&1] safe to
        // overwrite again only after the t+1 barrier, which program order gives us.
    }
}

__device__ __forceinline__ float edge_score(float l, float r, float w) {
    float x = l + r;
    float lr = (x > 0.f) ? x : NEG_SLOPE * x;
    return w * lr;
}

// ---------------- pass 1: segment max via int-key atomicMax ----------------
__global__ void gat_edge_max(const int* __restrict__ src, const int* __restrict__ dst,
                             const float* __restrict__ ew,
                             const float* __restrict__ el, const float* __restrict__ er,
                             int* __restrict__ emax, int nedges) {
    int e = blockIdx.x * blockDim.x + threadIdx.x;
    if (e >= nedges) return;
    int s = src[e], d = dst[e];
    float w = ew[e];
    const float4 l4 = *(const float4*)(el + (size_t)s * 4);
    const float4 r4 = *(const float4*)(er + (size_t)d * 4);
    float ls[4] = {l4.x, l4.y, l4.z, l4.w};
    float rs[4] = {r4.x, r4.y, r4.z, r4.w};
    #pragma unroll
    for (int h = 0; h < 4; ++h)
        atomicMax(&emax[d * 4 + h], orderedKey(edge_score(ls[h], rs[h], w)));
}

// ---------------- pass 2: denom = segment_sum(exp(e - emax[dst])) ----------------
__global__ void gat_edge_denom(const int* __restrict__ src, const int* __restrict__ dst,
                               const float* __restrict__ ew,
                               const float* __restrict__ el, const float* __restrict__ er,
                               const int* __restrict__ emax, float* __restrict__ denom,
                               int nedges) {
    int e = blockIdx.x * blockDim.x + threadIdx.x;
    if (e >= nedges) return;
    int s = src[e], d = dst[e];
    float w = ew[e];
    const float4 l4 = *(const float4*)(el + (size_t)s * 4);
    const float4 r4 = *(const float4*)(er + (size_t)d * 4);
    float ls[4] = {l4.x, l4.y, l4.z, l4.w};
    float rs[4] = {r4.x, r4.y, r4.z, r4.w};
    #pragma unroll
    for (int h = 0; h < 4; ++h) {
        float sc = edge_score(ls[h], rs[h], w);
        float mx = keyToFloat(emax[d * 4 + h]);
        atomicAdd(&denom[d * 4 + h], __expf(sc - mx));
    }
}

// ---------------- pass 3: out[dst] += a * ft[src] (64 lanes per edge) ----------------
__global__ void gat_edge_scatter(const int* __restrict__ src, const int* __restrict__ dst,
                                 const float* __restrict__ ew,
                                 const float* __restrict__ el, const float* __restrict__ er,
                                 const int* __restrict__ emax, const float* __restrict__ denom,
                                 const float* __restrict__ ft, float* __restrict__ out,
                                 int nedges) {
    int t = blockIdx.x * blockDim.x + threadIdx.x;   // nedges*64 = 51.2M < 2^31
    int e = t >> 6;
    if (e >= nedges) return;
    int idx = t & 63;           // idx = h*16 + d
    int h = idx >> 4;
    int s = src[e], d = dst[e];
    float sc = edge_score(el[s * 4 + h], er[d * 4 + h], ew[e]);
    float mx = keyToFloat(emax[d * 4 + h]);
    float a  = __expf(sc - mx) / denom[d * 4 + h];
    atomicAdd(&out[(size_t)d * HD + idx], ft[(size_t)s * HD + idx] * a);
}

extern "C" void kernel_launch(void* const* d_in, const int* in_sizes, int n_in,
                              void* d_out, int out_size, void* d_ws, size_t ws_size,
                              hipStream_t stream) {
    const float* feat   = (const float*)d_in[0];
    const int*   src    = (const int*)d_in[1];
    const int*   dst    = (const int*)d_in[2];
    const float* ew     = (const float*)d_in[3];
    const float* W      = (const float*)d_in[4];
    const float* attn_l = (const float*)d_in[5];
    const float* attn_r = (const float*)d_in[6];
    float* out = (float*)d_out;

    const int nnodes = in_sizes[0] / IN_FEATS;
    const int nedges = in_sizes[1];

    // workspace carve-up (floats): ft | el | er | emax(int) | denom  (~16 MB total)
    float* ft    = (float*)d_ws;
    float* el    = ft + (size_t)nnodes * HD;
    float* er    = el + (size_t)nnodes * HEADS;
    int*   emax  = (int*)(er + (size_t)nnodes * HEADS);
    float* denom = (float*)(emax + (size_t)nnodes * HEADS);

    const int nh   = nnodes * HEADS;
    const int outN = nnodes * HD;

    gat_init<<<(outN + 255) / 256, 256, 0, stream>>>(out, outN, emax, denom, nh);

    gat_gemm_wmma<<<(nnodes + MT * 16 - 1) / (MT * 16), 128, 0, stream>>>(
        feat, W, attn_l, attn_r, ft, el, er, nnodes);

    gat_edge_max<<<(nedges + 255) / 256, 256, 0, stream>>>(src, dst, ew, el, er, emax, nedges);

    gat_edge_denom<<<(nedges + 255) / 256, 256, 0, stream>>>(src, dst, ew, el, er, emax, denom, nedges);

    long long tot = (long long)nedges * 64;
    gat_edge_scatter<<<(int)((tot + 255) / 256), 256, 0, stream>>>(
        src, dst, ew, el, er, emax, denom, ft, out, nedges);
}
